// Retain_pertime_45019847196960
// MI455X (gfx1250) — matrix operand order repaired and checked by reading, not compiled, not run on hardware
//
#include <hip/hip_runtime.h>
#include <hip/hip_bf16.h>

// ---------------------------------------------------------------------------
// RETAIN-style model on gfx1250 (MI455X), wave32 + WMMA bf16 (f32 accumulate)
// + Tensor Data Mover for weight-tile staging into LDS.
//
// Sizes: B=32, T=64, P=63, I=E=H=256, G=3H=768.
// ---------------------------------------------------------------------------

typedef __attribute__((ext_vector_type(16))) __bf16 v16bf;
typedef __attribute__((ext_vector_type(8)))  float  v8f;
typedef __attribute__((ext_vector_type(4)))  unsigned int u32x4;
typedef __attribute__((ext_vector_type(8)))  int i32x8;
typedef __attribute__((ext_vector_type(4)))  int i32x4;

#if __has_builtin(__builtin_amdgcn_tensor_load_to_lds)
#define USE_TDM 1
#else
#define USE_TDM 0
#endif

static __device__ inline unsigned short bf16u(float f) {
  unsigned u = __builtin_bit_cast(unsigned, f);
  unsigned r = u + 0x7FFFu + ((u >> 16) & 1u);   // round-to-nearest-even
  return (unsigned short)(r >> 16);
}
static __device__ inline float bf2f(unsigned short h) {
  unsigned u = ((unsigned)h) << 16;
  return __builtin_bit_cast(float, u);
}
static __device__ inline v8f vzero8() {
  v8f z;
#pragma unroll
  for (int i = 0; i < 8; ++i) z[i] = 0.0f;
  return z;
}

// A fragment (16x32 bf16, MxK): lane l -> row m0+(l&15); per VGPR v:
// k = (l>=16 ? 16 : 0) + v*2 + {0,1}.
static __device__ inline v16bf load_a_frag(const unsigned short* As, int lda,
                                           int m0, int k0, int lane) {
  v16bf a;
  const unsigned short* row = As + (size_t)(m0 + (lane & 15)) * lda + k0;
  const int kh = (lane & 16) ? 16 : 0;
#pragma unroll
  for (int v = 0; v < 8; ++v) {
    int k = kh + (v << 1);
    unsigned pr = *(const unsigned*)(row + k);
    a[2 * v]     = __builtin_bit_cast(__bf16, (unsigned short)(pr & 0xFFFFu));
    a[2 * v + 1] = __builtin_bit_cast(__bf16, (unsigned short)(pr >> 16));
  }
  return a;
}

// B fragment (32x16 bf16, KxN): Bs layout [n][32] k-contiguous per column.
static __device__ inline v16bf load_b_frag(const unsigned short* Bs, int n0,
                                           int lane) {
  const unsigned short* pp =
      Bs + (size_t)(n0 + (lane & 15)) * 32 + ((lane & 16) ? 16 : 0);
  return *(const v16bf*)pp;   // 32B -> 2x ds_load_b128
}

static __device__ inline v8f wmma_bf16(v16bf a, v16bf b, v8f c) {
  return __builtin_amdgcn_wmma_f32_16x16x32_bf16(false, a, false, b, (short)0,
                                                 c, false, false);
}

#if USE_TDM
// TDM 2D tile load: bf16 tensor, tile (row_elems x rows), row stride
// stride_elems, destination LDS offset lds_off (Bs[n][row_elems] layout).
// D# field packing per CDNA5 ISA.
static __device__ inline void tdm_load_2d_bf16(unsigned lds_off,
                                               const void* gptr,
                                               unsigned rows,
                                               unsigned row_elems,
                                               unsigned tensor_d0,
                                               unsigned stride_elems) {
  unsigned long long ga = (unsigned long long)(size_t)gptr;
  u32x4 g0;
  g0[0] = 1u;                                   // count=1, user mode
  g0[1] = lds_off;                              // lds_addr (bytes)
  g0[2] = (unsigned)(ga & 0xFFFFFFFFu);         // global_addr[31:0]
  g0[3] = (unsigned)((ga >> 32) & 0x01FFFFFFu)  // global_addr[56:32]
          | 0x80000000u;                        // type=2 ("image")
  i32x8 g1;
  g1[0] = 0x00010000;                           // data_size=1 (2 bytes)
  g1[1] = (int)((tensor_d0 & 0xFFFFu) << 16);   // tensor_dim0[15:0]
  g1[2] = (int)(((tensor_d0 >> 16) & 0xFFFFu) | ((rows & 0xFFFFu) << 16));
  g1[3] = (int)(((rows >> 16) & 0xFFFFu) | ((row_elems & 0xFFFFu) << 16));
  g1[4] = (int)(rows & 0xFFFFu);                // tile_dim1 (tile_dim2=0)
  g1[5] = (int)stride_elems;                    // tensor_dim0_stride[31:0]
  g1[6] = 0;
  g1[7] = 0;
  i32x4 z4;
  z4[0] = 0; z4[1] = 0; z4[2] = 0; z4[3] = 0;
#if __clang_major__ >= 23
  i32x8 z8;
#pragma unroll
  for (int i = 0; i < 8; ++i) z8[i] = 0;
  __builtin_amdgcn_tensor_load_to_lds(g0, g1, z4, z4, z8, 0);
#else
  __builtin_amdgcn_tensor_load_to_lds(g0, g1, z4, z4, 0);
#endif
  __builtin_amdgcn_s_wait_tensorcnt(0);
}
#endif

// LDS byte offset of a __shared__ object: generic LDS addresses map to LDS
// by truncation to 32 bits (ISA 10.2 aperture rules).
static __device__ inline unsigned lds_offset(const void* p) {
  return (unsigned)(size_t)p;
}

// ---------------------------------------------------------------------------
// Elementwise prep kernels
// ---------------------------------------------------------------------------
__global__ __launch_bounds__(256) void k_cast_xt(const float* __restrict__ x,
                                                 unsigned short* __restrict__ xt) {
  int idx = blockIdx.x * 256 + threadIdx.x;    // 64*32*256
  int i = idx & 255;
  int rb = idx >> 8;                            // t*32+b
  int b = rb & 31;
  int t = rb >> 5;
  xt[idx] = bf16u(x[((size_t)b * 64 + t) * 256 + i]);
}

__global__ __launch_bounds__(256) void k_cast(const float* __restrict__ s,
                                              unsigned short* __restrict__ d,
                                              int n) {
  int idx = blockIdx.x * 256 + threadIdx.x;
  if (idx < n) d[idx] = bf16u(s[idx]);
}

__global__ __launch_bounds__(256) void k_make_w2t(const float* __restrict__ emb_w,
                                                  const float* __restrict__ out_w,
                                                  unsigned short* __restrict__ w2t) {
  int idx = blockIdx.x * 256 + threadIdx.x;    // 256*256
  int i = idx >> 8, e = idx & 255;
  w2t[(size_t)i * 256 + e] = bf16u(out_w[e] * emb_w[(size_t)e * 256 + i]);
}

// ---------------------------------------------------------------------------
// Generic GEMM: C[M,N] = A[M,K](bf16) @ W[N,K](f32)^T + bias[N]
// Block tile 64x256, 8 waves; grid = (M/64, N/256)
// ---------------------------------------------------------------------------
__global__ __launch_bounds__(256) void k_gemm_nt(
    const unsigned short* __restrict__ A, const float* __restrict__ W,
    const float* __restrict__ bias, float* __restrict__ Cf,
    unsigned short* __restrict__ Cb, int N, int K) {
  __shared__ alignas(32) unsigned short As[64 * 32];
  __shared__ alignas(32) unsigned short Bs[256 * 32];
  const int m_blk = blockIdx.x * 64;
  const int n_blk = blockIdx.y * 256;
  const int tid = threadIdx.x, lane = tid & 31, wave = tid >> 5;
  const int wm = (wave & 3) * 16;
  const int wn = (wave >> 2) * 128;

  v8f acc[8];
#pragma unroll
  for (int j = 0; j < 8; ++j) acc[j] = vzero8();

  for (int kc = 0; kc < K; kc += 32) {
    __syncthreads();
    // A chunk: 64x32 bf16 = 256 uint4, one per thread
    {
      int m = tid >> 2, q = tid & 3;
      *(uint4*)(As + (size_t)m * 32 + q * 8) =
          *(const uint4*)(A + (size_t)(m_blk + m) * K + kc + q * 8);
    }
    // W chunk: 256x32 f32 -> bf16, 2048 float4 / 8 per thread
    for (int i = tid; i < 2048; i += 256) {
      int nl = i >> 3, q = i & 7;
      float4 w4 = *(const float4*)(W + (size_t)(n_blk + nl) * K + kc + q * 4);
      uint2 pk;
      pk.x = (unsigned)bf16u(w4.x) | ((unsigned)bf16u(w4.y) << 16);
      pk.y = (unsigned)bf16u(w4.z) | ((unsigned)bf16u(w4.w) << 16);
      *(uint2*)(Bs + (size_t)nl * 32 + q * 4) = pk;
    }
    if (kc + 32 < K)
      __builtin_prefetch(W + (size_t)n_blk * K + kc + 32, 0, 1);
    __syncthreads();
    v16bf af = load_a_frag(As, 32, wm, 0, lane);
#pragma unroll
    for (int j = 0; j < 8; ++j) {
      v16bf bfr = load_b_frag(Bs, wn + j * 16, lane);
      acc[j] = wmma_bf16(af, bfr, acc[j]);
    }
  }
#pragma unroll
  for (int j = 0; j < 8; ++j) {
    int n = n_blk + wn + j * 16 + (lane & 15);
    float bv = bias[n];
#pragma unroll
    for (int v = 0; v < 8; ++v) {
      int m = m_blk + wm + ((lane & 16) ? 8 : 0) + v;
      float val = acc[j][v] + bv;
      if (Cf) Cf[(size_t)m * N + n] = val;
      if (Cb) Cb[(size_t)m * N + n] = bf16u(val);
    }
  }
}

// ---------------------------------------------------------------------------
// GRU: grid (63, 2).  Block p runs steps k=0..p for rows (p, b=0..31).
// h (32x256) resident in LDS; per step gh = h @ w_hh^T (32x768x256 WMMA),
// w_hh streamed L2 -> LDS in 768x32 tiles via the Tensor Data Mover.
// Wave w owns gate columns [w*32, w*32+32) for r/z/n -> in-register gates.
// ---------------------------------------------------------------------------
__global__ __launch_bounds__(256) void k_gru(
    const unsigned short* __restrict__ whh_a,
    const unsigned short* __restrict__ whh_b, const float* __restrict__ gxa,
    const float* __restrict__ gxb, const float* __restrict__ bhh_a,
    const float* __restrict__ bhh_b, unsigned short* __restrict__ ha,
    unsigned short* __restrict__ hb) {
  const int p = blockIdx.x;
  const int sel = blockIdx.y;
  const unsigned short* whh = sel ? whh_b : whh_a;
  const float* gx = sel ? gxb : gxa;
  const float* bhh = sel ? bhh_b : bhh_a;
  unsigned short* hout = sel ? hb : ha;

  __shared__ alignas(128) unsigned char smem[768 * 32 * 2 + 32 * 256 * 4 +
                                             32 * 256 * 2];
  unsigned short* Bs = (unsigned short*)smem;                    // 48KB
  float* h32 = (float*)(smem + 768 * 32 * 2);                    // 32KB
  unsigned short* h16 =
      (unsigned short*)(smem + 768 * 32 * 2 + 32 * 256 * 4);     // 16KB

  const int tid = threadIdx.x, lane = tid & 31, wave = tid >> 5;
  for (int i = tid; i < 32 * 256; i += 256) { h32[i] = 0.0f; h16[i] = 0; }

  float bg[3][2];
#pragma unroll
  for (int g = 0; g < 3; ++g)
#pragma unroll
    for (int nt = 0; nt < 2; ++nt)
      bg[g][nt] = bhh[g * 256 + wave * 32 + nt * 16 + (lane & 15)];

  const unsigned bs_lds = lds_offset(Bs);

  for (int k = 0; k <= p; ++k) {
    const int t = p - k;   // reversed-prefix input row for this step
    v8f acc[3][2][2];      // [gate][ntile][mtile]
#pragma unroll
    for (int g = 0; g < 3; ++g)
#pragma unroll
      for (int nt = 0; nt < 2; ++nt)
#pragma unroll
        for (int mt = 0; mt < 2; ++mt) acc[g][nt][mt] = vzero8();

    for (int kc = 0; kc < 256; kc += 32) {
      __syncthreads();   // prev chunk reads / prev-step h writes done
#if USE_TDM
      if (wave == 0)
        tdm_load_2d_bf16(bs_lds, whh + kc, 768u, 32u, 256u, 256u);
#else
      for (int i = tid; i < 3072; i += 256) {
        int nl = i >> 2, q = i & 3;
        *(uint4*)(Bs + (size_t)nl * 32 + q * 8) =
            *(const uint4*)(whh + (size_t)nl * 256 + kc + q * 8);
      }
#endif
      __syncthreads();
      v16bf a0 = load_a_frag(h16, 256, 0, kc, lane);
      v16bf a1 = load_a_frag(h16, 256, 16, kc, lane);
#pragma unroll
      for (int g = 0; g < 3; ++g) {
#pragma unroll
        for (int nt = 0; nt < 2; ++nt) {
          v16bf bfr = load_b_frag(Bs, g * 256 + wave * 32 + nt * 16, lane);
          acc[g][nt][0] = wmma_bf16(a0, bfr, acc[g][nt][0]);
          acc[g][nt][1] = wmma_bf16(a1, bfr, acc[g][nt][1]);
        }
      }
    }
    __syncthreads();   // all WMMA reads of h16 done before overwriting h
#pragma unroll
    for (int mt = 0; mt < 2; ++mt) {
#pragma unroll
      for (int nt = 0; nt < 2; ++nt) {
        const int c = wave * 32 + nt * 16 + (lane & 15);
#pragma unroll
        for (int v = 0; v < 8; ++v) {
          const int b = mt * 16 + ((lane & 16) ? 8 : 0) + v;
          const float* gxp = gx + ((size_t)t * 32 + b) * 768;
          float xr = gxp[c], xz = gxp[256 + c], xn = gxp[512 + c];
          float r = 1.0f / (1.0f + __expf(-(xr + acc[0][nt][mt][v] + bg[0][nt])));
          float z = 1.0f / (1.0f + __expf(-(xz + acc[1][nt][mt][v] + bg[1][nt])));
          float nn = tanhf(xn + r * (acc[2][nt][mt][v] + bg[2][nt]));
          float hnew = (1.0f - z) * nn + z * h32[b * 256 + c];
          h32[b * 256 + c] = hnew;
          unsigned short u = bf16u(hnew);
          h16[b * 256 + c] = u;
          hout[(((size_t)p * 64 + k) * 32 + b) * 256 + c] = u;
        }
      }
    }
  }
}

// ---------------------------------------------------------------------------
// alpha: pre[t,b] = 0.5*<ha[p][p-t][b], alpha_w> + alpha_b; softmax over t<=p
// ---------------------------------------------------------------------------
__global__ __launch_bounds__(256) void k_alpha(
    const unsigned short* __restrict__ ha, const float* __restrict__ alpha_w,
    const float* __restrict__ alpha_b, float* __restrict__ alpha) {
  const int p = blockIdx.x;
  __shared__ float pre_s[64 * 32];
  __shared__ float aw[256];
  const int tid = threadIdx.x;
  aw[tid] = alpha_w[tid];
  __syncthreads();
  for (int pair = tid; pair < 64 * 32; pair += 256) {
    int t = pair >> 5, b = pair & 31;
    if (t <= p) {
      const unsigned short* hp =
          ha + (((size_t)p * 64 + (p - t)) * 32 + b) * 256;
      float acc = 0.0f;
      for (int h = 0; h < 256; h += 8) {
        uint4 u = *(const uint4*)(hp + h);
        acc += bf2f((unsigned short)(u.x & 0xFFFFu)) * aw[h + 0];
        acc += bf2f((unsigned short)(u.x >> 16)) * aw[h + 1];
        acc += bf2f((unsigned short)(u.y & 0xFFFFu)) * aw[h + 2];
        acc += bf2f((unsigned short)(u.y >> 16)) * aw[h + 3];
        acc += bf2f((unsigned short)(u.z & 0xFFFFu)) * aw[h + 4];
        acc += bf2f((unsigned short)(u.z >> 16)) * aw[h + 5];
        acc += bf2f((unsigned short)(u.w & 0xFFFFu)) * aw[h + 6];
        acc += bf2f((unsigned short)(u.w >> 16)) * aw[h + 7];
      }
      pre_s[pair] = 0.5f * acc + alpha_b[0];
    } else {
      pre_s[pair] = -1e30f;
    }
  }
  __syncthreads();
  if (tid < 32) {
    int b = tid;
    float mx = -1e30f;
    for (int t = 0; t <= p; ++t) mx = fmaxf(mx, pre_s[t * 32 + b]);
    float s = 0.0f;
    for (int t = 0; t <= p; ++t) s += __expf(pre_s[t * 32 + b] - mx);
    float inv = 1.0f / s;
    for (int t = 0; t <= p; ++t)
      alpha[((size_t)p * 64 + t) * 32 + b] = __expf(pre_s[t * 32 + b] - mx) * inv;
  }
}

// ---------------------------------------------------------------------------
// attn: per p, loop t=0..p.  beta = tanh(0.5*hb@beta_w^T + b); s = alpha*beta;
// c += s*emb; v = s @ W2t^T; g += v*x.  pred/weight written to d_out.
// ---------------------------------------------------------------------------
__global__ __launch_bounds__(256) void k_attn(
    const unsigned short* __restrict__ hb, const float* __restrict__ alpha,
    const unsigned short* __restrict__ beta_w_bf,
    const float* __restrict__ beta_b, const unsigned short* __restrict__ w2t,
    const float* __restrict__ emb, const float* __restrict__ x,
    const float* __restrict__ out_w, const float* __restrict__ out_b,
    float* __restrict__ outp) {
  const int p = blockIdx.x;
  __shared__ alignas(32) unsigned short hstage[32 * 256];
  __shared__ alignas(32) unsigned short sstage[32 * 256];
  __shared__ alignas(128) unsigned short Bs[256 * 32];
  __shared__ float c_acc[32 * 256];
  __shared__ float g_acc[32 * 256];
  const int tid = threadIdx.x, lane = tid & 31, wave = tid >> 5;
  for (int i = tid; i < 32 * 256; i += 256) { c_acc[i] = 0.0f; g_acc[i] = 0.0f; }
  __syncthreads();
  const unsigned bs_lds = lds_offset(Bs);

  for (int t = 0; t <= p; ++t) {
    const unsigned short* hsrc = hb + (((size_t)p * 64 + (p - t)) * 32) * 256;
    for (int i = tid; i < 1024; i += 256)
      ((uint4*)hstage)[i] = ((const uint4*)hsrc)[i];
    __syncthreads();

    // GEMM1: braw[32x256] = hstage @ beta_w^T
    v8f accb[2][2];
#pragma unroll
    for (int nt = 0; nt < 2; ++nt)
#pragma unroll
      for (int mt = 0; mt < 2; ++mt) accb[nt][mt] = vzero8();
    for (int kc = 0; kc < 256; kc += 32) {
      __syncthreads();
#if USE_TDM
      if (wave == 0)
        tdm_load_2d_bf16(bs_lds, beta_w_bf + kc, 256u, 32u, 256u, 256u);
#else
      for (int i = tid; i < 1024; i += 256) {
        int nl = i >> 2, q = i & 3;
        *(uint4*)(Bs + (size_t)nl * 32 + q * 8) =
            *(const uint4*)(beta_w_bf + (size_t)nl * 256 + kc + q * 8);
      }
#endif
      __syncthreads();
      v16bf a0 = load_a_frag(hstage, 256, 0, kc, lane);
      v16bf a1 = load_a_frag(hstage, 256, 16, kc, lane);
#pragma unroll
      for (int nt = 0; nt < 2; ++nt) {
        v16bf bfr = load_b_frag(Bs, wave * 32 + nt * 16, lane);
        accb[nt][0] = wmma_bf16(a0, bfr, accb[nt][0]);
        accb[nt][1] = wmma_bf16(a1, bfr, accb[nt][1]);
      }
    }
    __syncthreads();
    // beta -> s -> c accumulate; each (b,e) has a unique owner thread
#pragma unroll
    for (int nt = 0; nt < 2; ++nt) {
      const int e = wave * 32 + nt * 16 + (lane & 15);
#pragma unroll
      for (int mt = 0; mt < 2; ++mt) {
#pragma unroll
        for (int v = 0; v < 8; ++v) {
          int b = mt * 16 + ((lane & 16) ? 8 : 0) + v;
          float beta = tanhf(0.5f * accb[nt][mt][v] + beta_b[e]);
          float s = alpha[((size_t)p * 64 + t) * 32 + b] * beta;
          c_acc[b * 256 + e] += s * emb[((size_t)t * 32 + b) * 256 + e];
          sstage[b * 256 + e] = bf16u(s);
        }
      }
    }
    __syncthreads();

    // GEMM2: v[32x256(I)] = sstage @ W2t^T  (W2t[i][e] = w0[e]*emb_w[e][i])
    v8f accv[2][2];
#pragma unroll
    for (int nt = 0; nt < 2; ++nt)
#pragma unroll
      for (int mt = 0; mt < 2; ++mt) accv[nt][mt] = vzero8();
    for (int kc = 0; kc < 256; kc += 32) {
      __syncthreads();
#if USE_TDM
      if (wave == 0)
        tdm_load_2d_bf16(bs_lds, w2t + kc, 256u, 32u, 256u, 256u);
#else
      for (int i = tid; i < 1024; i += 256) {
        int nl = i >> 2, q = i & 3;
        *(uint4*)(Bs + (size_t)nl * 32 + q * 8) =
            *(const uint4*)(w2t + (size_t)nl * 256 + kc + q * 8);
      }
#endif
      __syncthreads();
      v16bf a0 = load_a_frag(sstage, 256, 0, kc, lane);
      v16bf a1 = load_a_frag(sstage, 256, 16, kc, lane);
#pragma unroll
      for (int nt = 0; nt < 2; ++nt) {
        v16bf bfr = load_b_frag(Bs, wave * 32 + nt * 16, lane);
        accv[nt][0] = wmma_bf16(a0, bfr, accv[nt][0]);
        accv[nt][1] = wmma_bf16(a1, bfr, accv[nt][1]);
      }
    }
    __syncthreads();
#pragma unroll
    for (int nt = 0; nt < 2; ++nt) {
      const int ii = wave * 32 + nt * 16 + (lane & 15);
#pragma unroll
      for (int mt = 0; mt < 2; ++mt) {
#pragma unroll
        for (int v = 0; v < 8; ++v) {
          int b = mt * 16 + ((lane & 16) ? 8 : 0) + v;
          g_acc[b * 256 + ii] +=
              accv[nt][mt][v] * x[((size_t)b * 64 + t) * 256 + ii];
        }
      }
    }
    __syncthreads();
  }

  // pred[b,p] and weight[b,p,i]
  if (tid < 32) {
    int b = tid;
    float s = 0.0f;
    for (int e = 0; e < 256; ++e) s += c_acc[b * 256 + e] * out_w[e];
    outp[(size_t)b * 63 + p] = s + out_b[0];
  }
  float inv = 1.0f / (float)(p + 1);
  for (int i = tid; i < 32 * 256; i += 256) {
    int b = i >> 8, ii = i & 255;
    outp[2016 + ((size_t)b * 63 + p) * 256 + ii] = g_acc[i] * inv;
  }
}

// ---------------------------------------------------------------------------
extern "C" void kernel_launch(void* const* d_in, const int* in_sizes, int n_in,
                              void* d_out, int out_size, void* d_ws,
                              size_t ws_size, hipStream_t stream) {
  const float* x       = (const float*)d_in[0];
  const float* emb_w   = (const float*)d_in[1];
  const float* emb_b   = (const float*)d_in[2];
  const float* a_wih   = (const float*)d_in[3];
  const float* a_whh   = (const float*)d_in[4];
  const float* a_bih   = (const float*)d_in[5];
  const float* a_bhh   = (const float*)d_in[6];
  const float* b_wih   = (const float*)d_in[7];
  const float* b_whh   = (const float*)d_in[8];
  const float* b_bih   = (const float*)d_in[9];
  const float* b_bhh   = (const float*)d_in[10];
  const float* alpha_w = (const float*)d_in[11];
  const float* alpha_b = (const float*)d_in[12];
  const float* beta_w  = (const float*)d_in[13];
  const float* beta_b  = (const float*)d_in[14];
  const float* out_w   = (const float*)d_in[15];
  const float* out_b   = (const float*)d_in[16];
  float* outp = (float*)d_out;

  char* base = (char*)d_ws;
  size_t off = 0;
  auto alloc = [&](size_t bytes) -> char* {
    char* pp = base + off;
    off = (off + bytes + 255) & ~(size_t)255;
    return pp;
  };
  unsigned short* xt_bf  = (unsigned short*)alloc((size_t)2048 * 256 * 2);
  float*          embf   = (float*)alloc((size_t)2048 * 256 * 4);
  unsigned short* embb   = (unsigned short*)alloc((size_t)2048 * 256 * 2);
  float*          gxa    = (float*)alloc((size_t)2048 * 768 * 4);
  float*          gxb    = (float*)alloc((size_t)2048 * 768 * 4);
  unsigned short* whha   = (unsigned short*)alloc((size_t)768 * 256 * 2);
  unsigned short* whhb   = (unsigned short*)alloc((size_t)768 * 256 * 2);
  unsigned short* betawb = (unsigned short*)alloc((size_t)256 * 256 * 2);
  unsigned short* w2t    = (unsigned short*)alloc((size_t)256 * 256 * 2);
  unsigned short* ha     = (unsigned short*)alloc((size_t)63 * 64 * 32 * 256 * 2);
  unsigned short* hbuf   = (unsigned short*)alloc((size_t)63 * 64 * 32 * 256 * 2);
  float*          alpha  = (float*)alloc((size_t)63 * 64 * 32 * 4);

  k_cast_xt<<<2048, 256, 0, stream>>>(x, xt_bf);
  k_cast<<<768, 256, 0, stream>>>(a_whh, whha, 768 * 256);
  k_cast<<<768, 256, 0, stream>>>(b_whh, whhb, 768 * 256);
  k_cast<<<256, 256, 0, stream>>>(beta_w, betawb, 256 * 256);
  k_make_w2t<<<256, 256, 0, stream>>>(emb_w, out_w, w2t);

  k_gemm_nt<<<dim3(32, 1), 256, 0, stream>>>(xt_bf, emb_w, emb_b, embf, embb,
                                             256, 256);
  k_gemm_nt<<<dim3(32, 3), 256, 0, stream>>>(embb, a_wih, a_bih, gxa,
                                             (unsigned short*)nullptr, 768, 256);
  k_gemm_nt<<<dim3(32, 3), 256, 0, stream>>>(embb, b_wih, b_bih, gxb,
                                             (unsigned short*)nullptr, 768, 256);

  k_gru<<<dim3(63, 2), 256, 0, stream>>>(whha, whhb, gxa, gxb, a_bhh, b_bhh,
                                         ha, hbuf);
  k_alpha<<<63, 256, 0, stream>>>(ha, alpha_w, alpha_b, alpha);
  k_attn<<<63, 256, 0, stream>>>(hbuf, alpha, betawb, beta_b, w2t, embf, x,
                                 out_w, out_b, outp);
  (void)in_sizes; (void)n_in; (void)out_size; (void)ws_size;
}
